// PointNet2SemSeg_35407710388692
// MI455X (gfx1250) — matrix operand order repaired and checked by reading, not compiled, not run on hardware
//
#include <hip/hip_runtime.h>
#include <math.h>

typedef __attribute__((ext_vector_type(16))) _Float16 v16h;
typedef __attribute__((ext_vector_type(8)))  _Float16 v8h;
typedef __attribute__((ext_vector_type(8)))  float    v8f;

// ===========================================================================
// Farthest point sampling: one block per batch; iterative block-wide argmax.
// Matches JAX semantics: record `farthest` before the dist update; argmax
// returns first (lowest-index) occurrence of the max.
// ===========================================================================
#define FPS_T 256
__global__ __launch_bounds__(FPS_T)
void k_fps(const float* __restrict__ xyz, float* __restrict__ dist,
           int* __restrict__ idx, int N, int npoint)
{
  const int b = blockIdx.x;
  const float* px = xyz + (size_t)b * N * 3;
  float* pd = dist + (size_t)b * N;
  int*   pi = idx  + (size_t)b * npoint;
  __shared__ float sv[FPS_T];
  __shared__ int   si[FPS_T];
  __shared__ float c3[3];
  for (int j = threadIdx.x; j < N; j += FPS_T) pd[j] = 1e10f;
  __syncthreads();
  int far = 0;
  for (int it = 0; it < npoint; ++it) {
    if (threadIdx.x == 0) {
      pi[it] = far;
      c3[0] = px[far*3+0]; c3[1] = px[far*3+1]; c3[2] = px[far*3+2];
    }
    __syncthreads();
    float cx = c3[0], cy = c3[1], cz = c3[2];
    float bv = -1.0f; int bi = 0x7fffffff;
    for (int j = threadIdx.x; j < N; j += FPS_T) {
      float dx = px[j*3+0]-cx, dy = px[j*3+1]-cy, dz = px[j*3+2]-cz;
      float d = dx*dx + dy*dy + dz*dz;
      float dj = pd[j];
      if (d < dj) { dj = d; pd[j] = dj; }
      if (dj > bv) { bv = dj; bi = j; }   // strict > keeps first max in-thread
    }
    sv[threadIdx.x] = bv; si[threadIdx.x] = bi;
    __syncthreads();
    for (int s = FPS_T/2; s > 0; s >>= 1) {
      if (threadIdx.x < s) {
        float ov = sv[threadIdx.x+s]; int oi = si[threadIdx.x+s];
        if (ov > sv[threadIdx.x] ||
            (ov == sv[threadIdx.x] && oi < si[threadIdx.x])) {
          sv[threadIdx.x] = ov; si[threadIdx.x] = oi;
        }
      }
      __syncthreads();
    }
    far = si[0];
    __syncthreads();
  }
}

// gather 3-float coordinates by index: out[b,s,:] = src[b, idx[b,s], :]
__global__ void k_gather3(const float* __restrict__ src, const int* __restrict__ idx,
                          float* __restrict__ out, int B, int N, int S)
{
  int t = blockIdx.x * blockDim.x + threadIdx.x;
  if (t >= B * S) return;
  int b = t / S;
  int j = idx[t];
  const float* p = src + ((size_t)b * N + j) * 3;
  float* o = out + (size_t)t * 3;
  o[0] = p[0]; o[1] = p[1]; o[2] = p[2];
}

// Ball query: first K point-indices (ascending) within radius; pad with first.
__global__ void k_ball(const float* __restrict__ xyz, const float* __restrict__ nxyz,
                       int* __restrict__ gidx, int B, int N, int S, int K, float r2)
{
  int t = blockIdx.x * blockDim.x + threadIdx.x;
  if (t >= B * S) return;
  int b = t / S;
  const float* px = xyz + (size_t)b * N * 3;
  const float* q = nxyz + (size_t)t * 3;
  float qx = q[0], qy = q[1], qz = q[2];
  int* out = gidx + (size_t)t * K;
  int cnt = 0, first = 0;
  for (int j = 0; j < N && cnt < K; ++j) {
    float dx = px[j*3+0]-qx, dy = px[j*3+1]-qy, dz = px[j*3+2]-qz;
    if (dx*dx + dy*dy + dz*dz <= r2) {
      if (cnt == 0) first = j;
      out[cnt++] = j;
    }
  }
  for (; cnt < K; ++cnt) out[cnt] = first;
}

// Build grouped-feature rows. order==0: [points(Cp), gxyz] (MSG branch)
//                            order==1: [gxyz, points(Cp)] (SSG)
__global__ void k_group(const float* __restrict__ xyz, const float* __restrict__ pts,
                        const float* __restrict__ nxyz, const int* __restrict__ gidx,
                        float* __restrict__ out, int B, int N, int S, int K,
                        int Cp, int order)
{
  size_t t = (size_t)blockIdx.x * blockDim.x + threadIdx.x;
  size_t total = (size_t)B * S * K;
  if (t >= total) return;
  size_t bs = t / K;
  int b = (int)(bs / S);
  int j = gidx[t];
  const float* p = xyz + ((size_t)b * N + j) * 3;
  const float* q = nxyz + bs * 3;
  float gx = p[0]-q[0], gy = p[1]-q[1], gz = p[2]-q[2];
  const float* pp = pts + ((size_t)b * N + j) * Cp;
  int Cin = Cp + 3;
  float* o = out + t * Cin;
  if (order == 0) {
    for (int c = 0; c < Cp; ++c) o[c] = pp[c];
    o[Cp] = gx; o[Cp+1] = gy; o[Cp+2] = gz;
  } else {
    o[0] = gx; o[1] = gy; o[2] = gz;
    for (int c = 0; c < Cp; ++c) o[3+c] = pp[c];
  }
}

// ===========================================================================
// Padded f16 packing. Loads use clamped addresses + select (no exec branches).
// ===========================================================================
// X[M][Kd] f32 -> Xh[M][Kp] f16, zero pad for k >= Kd.  Kp % 32 == 0.
__global__ void k_pack_x(const float* __restrict__ X, _Float16* __restrict__ Xh,
                         int M, int Kd, int Kp)
{
  size_t t = (size_t)blockIdx.x * blockDim.x + threadIdx.x;
  size_t total = (size_t)M * Kp;
  if (t >= total) return;
  int k = (int)(t % Kp);
  size_t r = t / Kp;
  int ks = k < Kd ? k : Kd - 1;
  float v = X[r * Kd + ks];
  v = (k < Kd) ? v : 0.0f;
  Xh[t] = (_Float16)v;
}

// W[Kd][N] f32 -> Wt[Np][Kp] f16 transposed (B-fragment order), zero padded.
__global__ void k_pack_w(const float* __restrict__ W, _Float16* __restrict__ Wt,
                         int Kd, int N, int Kp, int Np)
{
  int t = blockIdx.x * blockDim.x + threadIdx.x;
  int total = Np * Kp;
  if (t >= total) return;
  int k = t % Kp, n = t / Kp;
  int ks = k < Kd ? k : Kd - 1;
  int ns = n < N ? n : N - 1;
  float v = W[(size_t)ks * N + ns];
  v = (k < Kd && n < N) ? v : 0.0f;
  Wt[t] = (_Float16)v;
}

// ===========================================================================
// WMMA GEMM on pre-packed padded f16 operands: Y[M,N] = Xh*Wt^T + bias.
// 128-thread block = 4 waves, each wave one 16x16 tile; block covers 64 rows.
// Requires M % 64 == 0 (true for every layer in this network), Kp % 32 == 0,
// Wt padded to Np rows.  No LDS, no bounds checks in the stream:
//   A frag: two contiguous 8-half runs at row, k offsets half*16 and half*16+8
//   B frag: 16 contiguous halves at row col of Wt, k offset half*16
//   C frag: elem r -> M r+8*half, col = l15        (C as in ISA 7.12.2; A/B k order contiguous)
// ===========================================================================
__global__ __launch_bounds__(128)
void k_wmma_gemm(const _Float16* __restrict__ Xh, const _Float16* __restrict__ Wt,
                 const float* __restrict__ bias, float* __restrict__ Y,
                 int M, int Kp, int N)
{
  const int wave = threadIdx.x >> 5;
  const int lane = threadIdx.x & 31;
  const int half = lane >> 4;
  const int l15  = lane & 15;
  const int mrow = blockIdx.x * 64 + wave * 16 + l15;
  const int col  = blockIdx.y * 16 + l15;
  const _Float16* ap = Xh + (size_t)mrow * Kp + half * 16;
  const _Float16* bp = Wt + (size_t)col  * Kp + half * 16;
  v8f acc = {};
  for (int kk = 0; kk < Kp; kk += 32) {
    v8h a0 = *(const v8h*)(ap + kk);        // K half*16 .. +7
    v8h a1 = *(const v8h*)(ap + kk + 8);    // K half*16+8 .. +7
    v16h b = *(const v16h*)(bp + kk);       // K e+16*half, e=0..15
    v16h a;
#pragma unroll
    for (int e = 0; e < 8; ++e) { a[e] = a0[e]; a[8 + e] = a1[e]; }
    acc = __builtin_amdgcn_wmma_f32_16x16x32_f16(
        /*neg_a=*/false, a, /*neg_b=*/false, b,
        /*c_mod=*/(short)0, acc, /*reuse_a=*/false, /*reuse_b=*/false);
  }
  if (col < N) {
    float bv = bias[col];
    const int mbase = blockIdx.x * 64 + wave * 16 + half * 8;
#pragma unroll
    for (int r = 0; r < 8; ++r)
      Y[(size_t)(mbase + r) * N + col] = acc[r] + bv;
  }
}

__global__ void k_zero(float* __restrict__ p, int n)
{
  int i = blockIdx.x * blockDim.x + threadIdx.x;
  if (i < n) p[i] = 0.0f;
}

// Per-channel sum / sumsq over a block's row range; register accumulation
// (coalesced across the wave per row), one global atomic per channel/block.
__global__ void k_bn_reduce(const float* __restrict__ Y, float* __restrict__ stats,
                            int M, int N, int rpb)
{
  int r0 = blockIdx.x * rpb;
  int r1 = r0 + rpb; if (r1 > M) r1 = M;
  for (int c = threadIdx.x; c < N; c += blockDim.x) {
    float s = 0.0f, q = 0.0f;
    for (int r = r0; r < r1; ++r) {
      float v = Y[(size_t)r * N + c];
      s += v; q += v * v;
    }
    atomicAdd(&stats[c], s);
    atomicAdd(&stats[N + c], q);
  }
}

// In-place BatchNorm (population var) + ReLU.
__global__ void k_bn_relu(float* __restrict__ Y, const float* __restrict__ stats,
                          const float* __restrict__ g, const float* __restrict__ be,
                          int M, int N, float inv)
{
  size_t i = (size_t)blockIdx.x * blockDim.x + threadIdx.x;
  size_t total = (size_t)M * N;
  if (i >= total) return;
  int c = (int)(i % N);
  float mean = stats[c] * inv;
  float var  = stats[N + c] * inv - mean * mean;
  float v = (Y[i] - mean) * rsqrtf(var + 1e-5f) * g[c] + be[c];
  Y[i] = v > 0.0f ? v : 0.0f;
}

// Max over the K (group) axis; scatter into concat buffer at channel offset.
__global__ void k_maxpool(const float* __restrict__ X, float* __restrict__ out,
                          int rows, int K, int C, int ostride, int ooff)
{
  int t = blockIdx.x * blockDim.x + threadIdx.x;
  if (t >= rows * C) return;
  int c = t % C, r = t / C;
  const float* p = X + ((size_t)r * K) * C + c;
  float m = p[0];
  for (int k = 1; k < K; ++k) m = fmaxf(m, p[(size_t)k * C]);
  out[(size_t)r * ostride + ooff + c] = m;
}

// Copy Csrc channels into a strided concat buffer at channel offset.
__global__ void k_copy_ch(const float* __restrict__ src, float* __restrict__ dst,
                          int rows, int C, int ostride, int ooff)
{
  int t = blockIdx.x * blockDim.x + threadIdx.x;
  if (t >= rows * C) return;
  int c = t % C, r = t / C;
  dst[(size_t)r * ostride + ooff + c] = src[(size_t)r * C + c];
}

// 3-NN inverse-distance interpolation (feature_propagation core).
__global__ void k_interp(const float* __restrict__ xyz1, const float* __restrict__ xyz2,
                         const float* __restrict__ pts2, float* __restrict__ out,
                         int B, int N1, int N2, int C2, int ostride, int ooff)
{
  int t = blockIdx.x * blockDim.x + threadIdx.x;
  if (t >= B * N1) return;
  int b = t / N1;
  const float* p1 = xyz1 + (size_t)t * 3;
  float x = p1[0], y = p1[1], z = p1[2];
  const float* p2 = xyz2 + (size_t)b * N2 * 3;
  float d0 = 1e30f, d1 = 1e30f, d2 = 1e30f;
  int i0 = 0, i1 = 0, i2 = 0;
  for (int j = 0; j < N2; ++j) {
    float dx = p2[j*3+0]-x, dy = p2[j*3+1]-y, dz = p2[j*3+2]-z;
    float d = dx*dx + dy*dy + dz*dz;
    if (d < d0)      { d2=d1; i2=i1; d1=d0; i1=i0; d0=d; i0=j; }
    else if (d < d1) { d2=d1; i2=i1; d1=d;  i1=j; }
    else if (d < d2) { d2=d;  i2=j; }
  }
  float w0 = 1.0f/(d0+1e-8f), w1 = 1.0f/(d1+1e-8f), w2 = 1.0f/(d2+1e-8f);
  float ws = w0 + w1 + w2;
  w0 /= ws; w1 /= ws; w2 /= ws;
  const float* f = pts2 + (size_t)b * N2 * C2;
  float* o = out + (size_t)t * ostride + ooff;
  for (int c = 0; c < C2; ++c)
    o[c] = w0 * f[(size_t)i0*C2 + c] + w1 * f[(size_t)i1*C2 + c]
         + w2 * f[(size_t)i2*C2 + c];
}

// sigmoid + (B,N,C) -> (B,C,N) transpose
__global__ void k_sig_tr(const float* __restrict__ X, float* __restrict__ out,
                         int B, int N, int C)
{
  size_t t = (size_t)blockIdx.x * blockDim.x + threadIdx.x;
  size_t total = (size_t)B * N * C;
  if (t >= total) return;
  int c = (int)(t % C);
  size_t bn = t / C;
  int n = (int)(bn % N);
  int b = (int)(bn / N);
  float v = X[t];
  out[((size_t)b * C + c) * N + n] = 1.0f / (1.0f + expf(-v));
}

// ===========================================================================
// Host orchestration
// ===========================================================================
extern "C" void kernel_launch(void* const* d_in, const int* in_sizes, int n_in,
                              void* d_out, int out_size, void* d_ws, size_t ws_size,
                              hipStream_t stream)
{
  (void)in_sizes; (void)n_in; (void)out_size;
  const int B = 4, N0 = 8192;
  const float* xyz   = (const float*)d_in[0];
  const float* color = (const float*)d_in[1];
  auto P = [&](int i) -> const float* { return (const float*)d_in[i]; };

  // Param base indices: params flattened depth-first in dict insertion order,
  // each layer dict contributing W, b, g, be.
  const int PI_SA05[3] = {2, 14, 26};
  const int PI_SA1 = 38, PI_SA2 = 50;
  const int PI_FP2 = 62, PI_FP1 = 70, PI_FP05 = 78;
  const int PI_HEAD = 86, PI_C2W = 90, PI_C2B = 91;

  // ---- workspace layout (256B aligned) ----
  size_t off = 0;
  auto A = [&](size_t bytes) -> size_t {
    size_t o = off; off += (bytes + 255) & ~(size_t)255; return o;
  };
  size_t o_dist   = A((size_t)B * 8192 * 4);
  size_t o_fps    = A((size_t)B * 2048 * 4);
  size_t o_l05xyz = A((size_t)B * 2048 * 3 * 4);
  size_t o_l1xyz  = A((size_t)B * 1024 * 3 * 4);
  size_t o_l2xyz  = A((size_t)B * 256  * 3 * 4);
  size_t o_gidx   = A((size_t)B * 2048 * 128 * 4);
  size_t o_l05p   = A((size_t)B * 2048 * 256 * 4);
  size_t o_l1p    = A((size_t)B * 1024 * 64  * 4);
  size_t o_l2p    = A((size_t)B * 256  * 128 * 4);
  size_t o_l1fp   = A((size_t)B * 1024 * 256 * 4);
  size_t o_l05fp  = A((size_t)B * 2048 * 128 * 4);
  size_t o_stats  = A((size_t)2 * 512 * 4);
  size_t o_xh     = A((size_t)1048576 * 96 * 2);    // packed f16 activations
  size_t o_wt     = A((size_t)512 * 512 * 2);       // packed f16 weights
  size_t o_bufA   = A((size_t)1048576 * 128 * 4);   // 512 MB
  size_t o_bufB   = A((size_t)1048576 * 128 * 4);   // 512 MB
  if (off > ws_size) return;  // workspace too small; bail deterministically

  char* ws = (char*)d_ws;
  float*    dist   = (float*)(ws + o_dist);
  int*      fpsI   = (int*)  (ws + o_fps);
  float*    l05xyz = (float*)(ws + o_l05xyz);
  float*    l1xyz  = (float*)(ws + o_l1xyz);
  float*    l2xyz  = (float*)(ws + o_l2xyz);
  int*      gidx   = (int*)  (ws + o_gidx);
  float*    l05p   = (float*)(ws + o_l05p);
  float*    l1p    = (float*)(ws + o_l1p);
  float*    l2p    = (float*)(ws + o_l2p);
  float*    l1fp   = (float*)(ws + o_l1fp);
  float*    l05fp  = (float*)(ws + o_l05fp);
  float*    stats  = (float*)(ws + o_stats);
  _Float16* xh     = (_Float16*)(ws + o_xh);
  _Float16* wt     = (_Float16*)(ws + o_wt);
  float*    bufA   = (float*)(ws + o_bufA);
  float*    bufB   = (float*)(ws + o_bufB);

  // GEMM with pre-packing: X[M][Cin] @ W[Cin][Cout] + b -> Y[M][Cout]
  auto gemm = [&](const float* X, const float* Wp, const float* bp, float* Y,
                  int M, int Cin, int Cout) {
    int Kp = (Cin + 31) & ~31;
    int Np = (Cout + 15) & ~15;
    size_t xtot = (size_t)M * Kp;
    k_pack_x<<<(unsigned)((xtot + 255) / 256), 256, 0, stream>>>(X, xh, M, Cin, Kp);
    k_pack_w<<<(Np * Kp + 255) / 256, 256, 0, stream>>>(Wp, wt, Cin, Cout, Kp, Np);
    dim3 g(M / 64, Np / 16);   // M % 64 == 0 for every layer in this network
    k_wmma_gemm<<<g, 128, 0, stream>>>(xh, wt, bp, Y, M, Kp, Cout);
  };

  // one MLP layer: GEMM (+bias) -> BN stats -> BN+ReLU (in-place on Y)
  auto mlp_layer = [&](const float* X, float* Y, int M, int Cin, int Cout, int pi) {
    gemm(X, P(pi), P(pi+1), Y, M, Cin, Cout);
    k_zero<<<(2*Cout + 255) / 256, 256, 0, stream>>>(stats, 2 * Cout);
    int rpb = 256;
    k_bn_reduce<<<(M + rpb - 1) / rpb, 256, 0, stream>>>(Y, stats, M, Cout, rpb);
    size_t tot = (size_t)M * Cout;
    k_bn_relu<<<(unsigned)((tot + 255) / 256), 256, 0, stream>>>(
        Y, stats, P(pi+2), P(pi+3), M, Cout, 1.0f / (float)M);
  };

  // ---------------- sa_msg (level 0.5): npoint=2048, 3 branches ----------------
  k_fps<<<B, FPS_T, 0, stream>>>(xyz, dist, fpsI, N0, 2048);
  k_gather3<<<(B*2048 + 255) / 256, 256, 0, stream>>>(xyz, fpsI, l05xyz, B, N0, 2048);

  const float brR2[3]  = {0.01f, 0.04f, 0.16f};
  const int   brK[3]   = {32, 64, 128};
  const int   brC[3][3] = {{32,32,64}, {64,64,64}, {64,96,128}};
  const int   brOff[3] = {0, 64, 128};
  for (int br = 0; br < 3; ++br) {
    const int S = 2048, K = brK[br];
    const int M = B * S * K;
    const int pi = PI_SA05[br];
    k_ball<<<(B*S + 255) / 256, 256, 0, stream>>>(xyz, l05xyz, gidx, B, N0, S, K, brR2[br]);
    k_group<<<(M + 255) / 256, 256, 0, stream>>>(xyz, color, l05xyz, gidx, bufA,
                                                 B, N0, S, K, 3, /*order msg*/0);
    mlp_layer(bufA, bufB, M, 6,          brC[br][0], pi);
    mlp_layer(bufB, bufA, M, brC[br][0], brC[br][1], pi + 4);
    mlp_layer(bufA, bufB, M, brC[br][1], brC[br][2], pi + 8);
    int t = B * S * brC[br][2];
    k_maxpool<<<(t + 255) / 256, 256, 0, stream>>>(bufB, l05p, B*S, K, brC[br][2],
                                                   256, brOff[br]);
  }

  // ---------------- sa_ssg level 1: 2048 -> 1024, r=0.2, K=32 ----------------
  k_fps<<<B, FPS_T, 0, stream>>>(l05xyz, dist, fpsI, 2048, 1024);
  k_gather3<<<(B*1024 + 255) / 256, 256, 0, stream>>>(l05xyz, fpsI, l1xyz, B, 2048, 1024);
  {
    const int S = 1024, K = 32, M = B * S * K;   // M = 131072, Cin = 259
    k_ball<<<(B*S + 255) / 256, 256, 0, stream>>>(l05xyz, l1xyz, gidx, B, 2048, S, K, 0.04f);
    k_group<<<(M + 255) / 256, 256, 0, stream>>>(l05xyz, l05p, l1xyz, gidx, bufA,
                                                 B, 2048, S, K, 256, /*order ssg*/1);
    mlp_layer(bufA, bufB, M, 259, 32, PI_SA1);
    mlp_layer(bufB, bufA, M, 32,  32, PI_SA1 + 4);
    mlp_layer(bufA, bufB, M, 32,  64, PI_SA1 + 8);
    int t = B * S * 64;
    k_maxpool<<<(t + 255) / 256, 256, 0, stream>>>(bufB, l1p, B*S, K, 64, 64, 0);
  }

  // ---------------- sa_ssg level 2: 1024 -> 256, r=0.4, K=32 ----------------
  k_fps<<<B, FPS_T, 0, stream>>>(l1xyz, dist, fpsI, 1024, 256);
  k_gather3<<<(B*256 + 255) / 256, 256, 0, stream>>>(l1xyz, fpsI, l2xyz, B, 1024, 256);
  {
    const int S = 256, K = 32, M = B * S * K;    // M = 32768, Cin = 67
    k_ball<<<(B*S + 255) / 256, 256, 0, stream>>>(l1xyz, l2xyz, gidx, B, 1024, S, K, 0.16f);
    k_group<<<(M + 255) / 256, 256, 0, stream>>>(l1xyz, l1p, l2xyz, gidx, bufA,
                                                 B, 1024, S, K, 64, 1);
    mlp_layer(bufA, bufB, M, 67, 64,  PI_SA2);
    mlp_layer(bufB, bufA, M, 64, 64,  PI_SA2 + 4);
    mlp_layer(bufA, bufB, M, 64, 128, PI_SA2 + 8);
    int t = B * S * 128;
    k_maxpool<<<(t + 255) / 256, 256, 0, stream>>>(bufB, l2p, B*S, K, 128, 128, 0);
  }

  // ---------------- fp2: (l1 <- l2), concat [l1_p(64), interp(128)] = 192 ----
  {
    const int rows = B * 1024;
    k_copy_ch<<<(rows*64 + 255) / 256, 256, 0, stream>>>(l1p, bufA, rows, 64, 192, 0);
    k_interp<<<(rows + 255) / 256, 256, 0, stream>>>(l1xyz, l2xyz, l2p, bufA,
                                                     B, 1024, 256, 128, 192, 64);
    mlp_layer(bufA, bufB, rows, 192, 512, PI_FP2);
    mlp_layer(bufB, bufA, rows, 512, 256, PI_FP2 + 4);
    k_copy_ch<<<(rows*256 + 255) / 256, 256, 0, stream>>>(bufA, l1fp, rows, 256, 256, 0);
  }

  // ---------------- fp1: (l05 <- l1), concat [l05_p(256), interp(256)] = 512 --
  {
    const int rows = B * 2048;
    k_copy_ch<<<(rows*256 + 255) / 256, 256, 0, stream>>>(l05p, bufA, rows, 256, 512, 0);
    k_interp<<<(rows + 255) / 256, 256, 0, stream>>>(l05xyz, l1xyz, l1fp, bufA,
                                                     B, 2048, 1024, 256, 512, 256);
    mlp_layer(bufA, bufB, rows, 512, 256, PI_FP1);
    mlp_layer(bufB, bufA, rows, 256, 128, PI_FP1 + 4);
    k_copy_ch<<<(rows*128 + 255) / 256, 256, 0, stream>>>(bufA, l05fp, rows, 128, 128, 0);
  }

  // ---------------- fp05: (l0 <- l05), concat [color(3), interp(128)] = 131 ---
  {
    const int rows = B * N0;   // 32768
    k_copy_ch<<<(rows*3 + 255) / 256, 256, 0, stream>>>(color, bufA, rows, 3, 131, 0);
    k_interp<<<(rows + 255) / 256, 256, 0, stream>>>(xyz, l05xyz, l05fp, bufA,
                                                     B, N0, 2048, 128, 131, 3);
    mlp_layer(bufA, bufB, rows, 131, 128, PI_FP05);
    mlp_layer(bufB, bufA, rows, 128, 64,  PI_FP05 + 4);

    // head1: 64 -> 128 (with BN+ReLU)
    mlp_layer(bufA, bufB, rows, 64, 128, PI_HEAD);

    // conv2: 128 -> 13 (+bias), then sigmoid + transpose to (B, 13, N)
    gemm(bufB, P(PI_C2W), P(PI_C2B), bufA, rows, 128, 13);
    size_t tot = (size_t)rows * 13;
    k_sig_tr<<<(unsigned)((tot + 255) / 256), 256, 0, stream>>>(
        bufA, (float*)d_out, B, N0, 13);
  }
}
